// LongformerClass_566935683128
// MI455X (gfx1250) — matrix operand order repaired and checked by reading, not compiled, not run on hardware
//
#include <hip/hip_runtime.h>
#include <math.h>

// ---------------------------------------------------------------------------
// Model constants (from reference setup_inputs)
// ---------------------------------------------------------------------------
#define BB 2
#define SS 4096
#define DD 768
#define FF 3072
#define LL 12
#define HH 12
#define DH 64
#define NL 10
#define NEGV (-1e9f)

typedef __attribute__((ext_vector_type(16))) __bf16 bf16x16;
typedef __attribute__((ext_vector_type(8)))  __bf16 bf16x8;
typedef __attribute__((ext_vector_type(8)))  float  f32x8;

union frag_u { bf16x16 f; bf16x8 h[2]; };

#define ACT_NONE 0
#define ACT_GELU 1
#define ACT_TANH 2

// A-fragment element->K mapping for 16-bit 16x16x32 WMMA (ISA 7.12.2):
// lane half 0 holds K {0..7,16..23}; half 1 holds K {8..15,24..31}.
// Storing tiles with octets 1<->2 swapped makes each lane's 16 elements
// contiguous at offset 16*half.
__device__ __forceinline__ int swap_oct(int o) {
    return (o == 1) ? 2 : (o == 2 ? 1 : o);
}

// load 8 consecutive floats, convert to 8 bf16 (two global_load_b128)
__device__ __forceinline__ bf16x8 ld8_bf16(const float* p) {
    const float4* q = (const float4*)p;
    float4 a = q[0], b = q[1];
    bf16x8 r;
    r[0] = (__bf16)a.x; r[1] = (__bf16)a.y; r[2] = (__bf16)a.z; r[3] = (__bf16)a.w;
    r[4] = (__bf16)b.x; r[5] = (__bf16)b.y; r[6] = (__bf16)b.z; r[7] = (__bf16)b.w;
    return r;
}

__device__ __forceinline__ bf16x8 zero8() {
    bf16x8 z;
    #pragma unroll
    for (int t = 0; t < 8; ++t) z[t] = (__bf16)0.f;
    return z;
}

__device__ __forceinline__ frag_u ld_frag(const __bf16* p) {
    frag_u u;
    u.h[0] = *(const bf16x8*)p;
    u.h[1] = *(const bf16x8*)(p + 8);
    return u;
}

// ---------------------------------------------------------------------------
// Generic GEMM: C[M,N] = act((A[M,K] @ W[K,N] + bias) * scale)
// Requires K % 32 == 0 (true for all calls: K in {768, 3072}).
// bf16 WMMA, f32 accum. 256 threads = 8 waves; block tile 128x64;
// wave tile 32x32 (4 WMMAs per K-step, 2 A-frags x 2 B-frags).
// Double-buffered LDS, one barrier per K-step, b128 fragment loads.
// ---------------------------------------------------------------------------
__global__ __launch_bounds__(256) void gemm_wmma_kernel(
    const float* __restrict__ A, long lda,
    const float* __restrict__ W, int ldw,
    const float* __restrict__ bias,
    float* __restrict__ C, long ldc,
    int M, int N, int K, float scale, int act)
{
    __shared__ __bf16 As[2][128][40];  // [buf][row][k linear], 80B row stride
    __shared__ __bf16 Bs[2][64][40];   // [buf][col n][k linear]

    const int tid  = threadIdx.x;
    const int lane = tid & 31;
    const int wave = tid >> 5;
    const int hf   = lane >> 4;
    const int lm   = lane & 15;
    const int wm   = wave >> 1;      // 0..3 : 32-row group
    const int wn   = wave & 1;       // 0..1 : 32-col group

    const long mBase = (long)blockIdx.y * 128;
    const long nBase = (long)blockIdx.x * 64;
    const int  m0 = wm * 32;
    const int  n0 = wn * 32;

    // staging thread mapping
    const int aOct = tid & 3;        // k octet 0..3
    const int aRow = tid >> 2;       // 0..63 (two row halves)
    const int bN4  = tid & 7;        // 4-col group 0..7 (per 32-col half)
    const int bKK  = tid >> 3;       // 0..31

    auto stageA = [&](int p, int k0) {
        #pragma unroll
        for (int rh = 0; rh < 2; ++rh) {
            int  rr  = aRow + rh * 64;
            long row = mBase + rr;
            bf16x8* dst = (bf16x8*)&As[p][rr][aOct * 8];
            if (row < M) *dst = ld8_bf16(&A[row * lda + k0 + aOct * 8]);
            else         *dst = zero8();
        }
    };
    auto stageB = [&](int p, int k0) {
        const int k = k0 + bKK;                  // always < K (K%32==0)
        #pragma unroll
        for (int nh = 0; nh < 2; ++nh) {
            int  nn  = nh * 32 + 4 * bN4;
            long col = nBase + nn;
            if (((ldw & 3) == 0) && (col + 3) < N) {
                float4 w4 = *(const float4*)&W[(long)k * ldw + col];
                Bs[p][nn + 0][bKK] = (__bf16)w4.x;
                Bs[p][nn + 1][bKK] = (__bf16)w4.y;
                Bs[p][nn + 2][bKK] = (__bf16)w4.z;
                Bs[p][nn + 3][bKK] = (__bf16)w4.w;
            } else {
                #pragma unroll
                for (int j = 0; j < 4; ++j) {
                    long cc = col + j;
                    float v = (cc < N) ? W[(long)k * ldw + cc] : 0.f;
                    Bs[p][nn + j][bKK] = (__bf16)v;
                }
            }
        }
    };

    f32x8 acc[2][2] = {};

    stageA(0, 0);
    stageB(0, 0);
    int p = 0;
    for (int k0 = 0; k0 < K; k0 += 32) {
        __syncthreads();                         // buffer p ready
        if (k0 + 32 < K) {                       // overlap next-tile loads
            stageA(p ^ 1, k0 + 32);
            stageB(p ^ 1, k0 + 32);
        }
        frag_u a0 = ld_frag(&As[p][m0 + lm][16 * hf]);
        frag_u a1 = ld_frag(&As[p][m0 + 16 + lm][16 * hf]);
        frag_u b0 = ld_frag(&Bs[p][n0 + lm][16 * hf]);
        frag_u b1 = ld_frag(&Bs[p][n0 + 16 + lm][16 * hf]);

        acc[0][0] = __builtin_amdgcn_wmma_f32_16x16x32_bf16(
            false, a0.f, false, b0.f, (short)0, acc[0][0], false, false);
        acc[0][1] = __builtin_amdgcn_wmma_f32_16x16x32_bf16(
            false, a0.f, false, b1.f, (short)0, acc[0][1], false, false);
        acc[1][0] = __builtin_amdgcn_wmma_f32_16x16x32_bf16(
            false, a1.f, false, b0.f, (short)0, acc[1][0], false, false);
        acc[1][1] = __builtin_amdgcn_wmma_f32_16x16x32_bf16(
            false, a1.f, false, b1.f, (short)0, acc[1][1], false, false);
        p ^= 1;
    }

    #pragma unroll
    for (int rt = 0; rt < 2; ++rt) {
        #pragma unroll
        for (int r = 0; r < 8; ++r) {
            long row = mBase + m0 + 16 * rt + r + 8 * hf;
            if (row >= M) continue;
            #pragma unroll
            for (int ct = 0; ct < 2; ++ct) {
                long col = nBase + n0 + 16 * ct + lm;
                if (col < N) {
                    float v = acc[rt][ct][r] + (bias ? bias[col] : 0.f);
                    v *= scale;
                    if (act == ACT_GELU)      v = 0.5f * v * (1.f + erff(v * 0.70710678118f));
                    else if (act == ACT_TANH) v = tanhf(v);
                    C[row * ldc + col] = v;
                }
            }
        }
    }
}

// ---------------------------------------------------------------------------
// Sliding-window attention (flash-style, WMMA). One wave per 16-query tile.
// Out-of-range / masked keys are killed by exp(NEG - m) == 0, so all loads
// use clamped addresses with no branches.
// ---------------------------------------------------------------------------
__global__ __launch_bounds__(32) void sliding_attn_kernel(
    const float* __restrict__ q, const float* __restrict__ k,
    const float* __restrict__ v, const int* __restrict__ mask,
    float* __restrict__ out)
{
    const int lane = threadIdx.x;
    const int hf = lane >> 4, lm = lane & 15;
    const int qt = blockIdx.x;
    const int h  = blockIdx.y;
    const int b  = blockIdx.z;
    const int q0 = qt * 16;
    const long base = ((long)b * SS) * DD + (long)h * DH;

    // Q fragments: per-lane elements are contiguous octet pairs -> b128 loads
    frag_u qfL, qfH;
    {
        const float* qp = &q[base + (long)(q0 + lm) * DD];
        qfL.h[0] = ld8_bf16(qp + 8 * hf);            // K {0..7}/{8..15}
        qfL.h[1] = ld8_bf16(qp + 16 + 8 * hf);       // K {16..23}/{24..31}
        qfH.h[0] = ld8_bf16(qp + 32 + 8 * hf);
        qfH.h[1] = ld8_bf16(qp + 48 + 8 * hf);
    }

    // global-token score per row: gs = q_row . k[pos 0]
    float gs = 0.f;
    {
        const float4* qp4 = (const float4*)&q[base + (long)(q0 + (lane & 15)) * DD];
        const float4* kp4 = (const float4*)&k[base];
        #pragma unroll
        for (int d = 0; d < 16; ++d) {
            float4 a = qp4[d], c = kp4[d];
            gs += a.x * c.x + a.y * c.y + a.z * c.z + a.w * c.w;
        }
    }

    float mrow[8], lrow[8];
    #pragma unroll
    for (int r = 0; r < 8; ++r) {
        mrow[r] = __shfl(gs, r + 8 * hf, 32);
        lrow[r] = 1.f;
    }
    // O fragments init with p_g(=1) * v[pos 0]
    f32x8 of[4];
    #pragma unroll
    for (int f = 0; f < 4; ++f) {
        float v0 = v[base + f * 16 + lm];
        #pragma unroll
        for (int r = 0; r < 8; ++r) of[f][r] = v0;
    }

    __shared__ __bf16 Ps[16][40];   // octet-permuted P tile

    for (int c = 0; c < 17; ++c) {               // 17*32 keys cover [q0-256, q0+15+256]
        const int ks = q0 - 256 + c * 32;

        // ---- QK^T : two 16-key tiles (B frag K is linear: contiguous loads)
        f32x8 st[2];
        #pragma unroll
        for (int t = 0; t < 2; ++t) {
            const int kpos = ks + t * 16 + lm;
            const int kc   = min(max(kpos, 0), SS - 1);
            const float* kp = &k[base + (long)kc * DD];
            frag_u kfL, kfH;
            kfL.h[0] = ld8_bf16(kp + 8 * hf);
            kfL.h[1] = ld8_bf16(kp + 16 + 8 * hf);
            kfH.h[0] = ld8_bf16(kp + 32 + 8 * hf);
            kfH.h[1] = ld8_bf16(kp + 48 + 8 * hf);

            f32x8 s = {};
            s = __builtin_amdgcn_wmma_f32_16x16x32_bf16(false, qfL.f, false, kfL.f, (short)0, s, false, false);
            s = __builtin_amdgcn_wmma_f32_16x16x32_bf16(false, qfH.f, false, kfH.f, (short)0, s, false, false);

            const bool kok = (kpos >= 1) && (kpos < SS) && (mask[(long)b * SS + kc] > 0);
            #pragma unroll
            for (int r = 0; r < 8; ++r) {
                int qpos = q0 + r + 8 * hf;
                bool ok = kok && (kpos >= qpos - 256) && (kpos <= qpos + 256);
                s[r] = ok ? s[r] : NEGV;
            }
            st[t] = s;
        }

        // ---- online softmax across chunk (row reductions within 16 lanes)
        float p0[8], p1[8], fac[8];
        #pragma unroll
        for (int r = 0; r < 8; ++r) {
            float mx = fmaxf(st[0][r], st[1][r]);
            #pragma unroll
            for (int off = 1; off < 16; off <<= 1)
                mx = fmaxf(mx, __shfl_xor(mx, off, 32));
            float nm = fmaxf(mrow[r], mx);
            fac[r] = __expf(mrow[r] - nm);
            p0[r]  = __expf(st[0][r] - nm);
            p1[r]  = __expf(st[1][r] - nm);
            float sm = p0[r] + p1[r];
            #pragma unroll
            for (int off = 1; off < 16; off <<= 1)
                sm += __shfl_xor(sm, off, 32);
            lrow[r] = lrow[r] * fac[r] + sm;
            mrow[r] = nm;
        }
        #pragma unroll
        for (int f = 0; f < 4; ++f)
            #pragma unroll
            for (int r = 0; r < 8; ++r) of[f][r] *= fac[r];

        // ---- P tile to LDS in octet-permuted order (A-fragment friendly)
        {
            int cbase = (lm & 7) + ((lm >> 3) << 4);
            #pragma unroll
            for (int r = 0; r < 8; ++r) {
                Ps[r + 8 * hf][cbase]     = (__bf16)p0[r];
                Ps[r + 8 * hf][cbase + 8] = (__bf16)p1[r];
            }
        }
        __syncthreads();
        frag_u pf = ld_frag(&Ps[lm][16 * hf]);

        // ---- PV: contraction over 32 keys, 4 output-dim tiles
        frag_u vf[4];
        #pragma unroll
        for (int i = 0; i < 16; ++i) {
            int kpos = ks + 8 * hf + i + (i >= 8 ? 8 : 0);
            int kc   = min(max(kpos, 0), SS - 1);
            const float* vp = &v[base + (long)kc * DD];
            #pragma unroll
            for (int f = 0; f < 4; ++f)
                vf[f].f[i] = (__bf16)vp[f * 16 + lm];
        }
        #pragma unroll
        for (int f = 0; f < 4; ++f)
            of[f] = __builtin_amdgcn_wmma_f32_16x16x32_bf16(
                false, pf.f, false, vf[f].f, (short)0, of[f], false, false);
        __syncthreads();
    }

    #pragma unroll
    for (int f = 0; f < 4; ++f)
        #pragma unroll
        for (int r = 0; r < 8; ++r) {
            int row = q0 + r + 8 * hf;
            out[base + (long)row * DD + f * 16 + lm] = of[f][r] / lrow[r];
        }
}

// ---------------------------------------------------------------------------
// Global-token attention: token 0 attends to all S with dedicated q/k/v.
// ---------------------------------------------------------------------------
__global__ __launch_bounds__(256) void global_attn_kernel(
    const float* __restrict__ qg, const float* __restrict__ kg,
    const float* __restrict__ vg, const int* __restrict__ mask,
    float* __restrict__ out)
{
    __shared__ float sc[SS];
    __shared__ float red[256];
    __shared__ float oacc[4][DH];
    __shared__ float qh[DH];

    const int h = blockIdx.x, b = blockIdx.y;
    const int tid = threadIdx.x;
    const long base = ((long)b * SS) * DD + (long)h * DH;

    if (tid < DH) qh[tid] = qg[(long)b * DD + h * DH + tid];
    __syncthreads();

    float lmax = -3.4e38f;
    for (int s = tid; s < SS; s += 256) {
        float a = 0.f;
        const float4* kp4 = (const float4*)&kg[base + (long)s * DD];
        const float4* q4  = (const float4*)qh;
        #pragma unroll
        for (int d = 0; d < 16; ++d) {
            float4 x = q4[d], y = kp4[d];
            a += x.x * y.x + x.y * y.y + x.z * y.z + x.w * y.w;
        }
        a += (mask[(long)b * SS + s] > 0) ? 0.f : NEGV;
        sc[s] = a;
        lmax = fmaxf(lmax, a);
    }
    red[tid] = lmax; __syncthreads();
    for (int o = 128; o > 0; o >>= 1) {
        if (tid < o) red[tid] = fmaxf(red[tid], red[tid + o]);
        __syncthreads();
    }
    float m = red[0]; __syncthreads();

    float ls = 0.f;
    for (int s = tid; s < SS; s += 256) {
        float p = __expf(sc[s] - m);
        sc[s] = p; ls += p;
    }
    red[tid] = ls; __syncthreads();
    for (int o = 128; o > 0; o >>= 1) {
        if (tid < o) red[tid] += red[tid + o];
        __syncthreads();
    }
    float denom = red[0];

    int d = tid & 63, part = tid >> 6;
    float acc = 0.f;
    for (int s = part; s < SS; s += 4) acc += sc[s] * vg[base + (long)s * DD + d];
    oacc[part][d] = acc; __syncthreads();
    if (tid < DH) {
        float ov = (oacc[0][tid] + oacc[1][tid] + oacc[2][tid] + oacc[3][tid]) / denom;
        out[base + tid] = ov;   // row s=0
    }
}

// ---------------------------------------------------------------------------
// Embedding + LayerNorm
// ---------------------------------------------------------------------------
__global__ __launch_bounds__(256) void embed_ln_kernel(
    const int* __restrict__ ids, const int* __restrict__ tt,
    const float* __restrict__ wordE, const float* __restrict__ posE,
    const float* __restrict__ typeE,
    const float* __restrict__ g, const float* __restrict__ be,
    float* __restrict__ h)
{
    const long row = blockIdx.x;
    const int s = (int)(row % SS);
    const int tid = threadIdx.x;
    __shared__ float red[256];
    __shared__ float xbuf[DD];

    long id = ids[row];
    long ty = tt[row];
    float lsum = 0.f;
    for (int d = tid; d < DD; d += 256) {
        float x = wordE[id * DD + d] + posE[(long)(s + 2) * DD + d] + typeE[ty * DD + d];
        xbuf[d] = x; lsum += x;
    }
    red[tid] = lsum; __syncthreads();
    for (int o = 128; o > 0; o >>= 1) { if (tid < o) red[tid] += red[tid + o]; __syncthreads(); }
    float mean = red[0] / DD; __syncthreads();
    float lvar = 0.f;
    for (int d = tid; d < DD; d += 256) { float t = xbuf[d] - mean; lvar += t * t; }
    red[tid] = lvar; __syncthreads();
    for (int o = 128; o > 0; o >>= 1) { if (tid < o) red[tid] += red[tid + o]; __syncthreads(); }
    float rstd = rsqrtf(red[0] / DD + 1e-5f);
    for (int d = tid; d < DD; d += 256)
        h[row * DD + d] = (xbuf[d] - mean) * rstd * g[d] + be[d];
}

// ---------------------------------------------------------------------------
// h = LN(h + y) * g + b   (in place on h)
// ---------------------------------------------------------------------------
__global__ __launch_bounds__(256) void add_ln_kernel(
    float* __restrict__ h, const float* __restrict__ y,
    const float* __restrict__ g, const float* __restrict__ be)
{
    const long row = blockIdx.x;
    const int tid = threadIdx.x;
    __shared__ float red[256];
    __shared__ float xbuf[DD];

    float lsum = 0.f;
    for (int d = tid; d < DD; d += 256) {
        float x = h[row * DD + d] + y[row * DD + d];
        xbuf[d] = x; lsum += x;
    }
    red[tid] = lsum; __syncthreads();
    for (int o = 128; o > 0; o >>= 1) { if (tid < o) red[tid] += red[tid + o]; __syncthreads(); }
    float mean = red[0] / DD; __syncthreads();
    float lvar = 0.f;
    for (int d = tid; d < DD; d += 256) { float t = xbuf[d] - mean; lvar += t * t; }
    red[tid] = lvar; __syncthreads();
    for (int o = 128; o > 0; o >>= 1) { if (tid < o) red[tid] += red[tid + o]; __syncthreads(); }
    float rstd = rsqrtf(red[0] / DD + 1e-5f);
    for (int d = tid; d < DD; d += 256)
        h[row * DD + d] = (xbuf[d] - mean) * rstd * g[d] + be[d];
}

// ---------------------------------------------------------------------------
// Host driver
// ---------------------------------------------------------------------------
static inline void launch_gemm(const float* A, long lda, const float* W, int ldw,
                               const float* bias, float* C, long ldc,
                               int M, int N, int K, float scale, int act,
                               hipStream_t stream)
{
    dim3 grid((N + 63) / 64, (M + 127) / 128);
    gemm_wmma_kernel<<<grid, 256, 0, stream>>>(A, lda, W, ldw, bias, C, ldc,
                                               M, N, K, scale, act);
}

extern "C" void kernel_launch(void* const* d_in, const int* in_sizes, int n_in,
                              void* d_out, int out_size, void* d_ws, size_t ws_size,
                              hipStream_t stream)
{
    (void)in_sizes; (void)n_in; (void)out_size; (void)ws_size;

    const int*   ids    = (const int*)d_in[0];
    const int*   mask   = (const int*)d_in[1];
    const int*   tti    = (const int*)d_in[2];
    const float* wordE  = (const float*)d_in[3];
    const float* posE   = (const float*)d_in[4];
    const float* typeE  = (const float*)d_in[5];
    const float* ln_e_g = (const float*)d_in[6];
    const float* ln_e_b = (const float*)d_in[7];
    const float* Wq  = (const float*)d_in[8];   const float* bq  = (const float*)d_in[9];
    const float* Wk  = (const float*)d_in[10];  const float* bk  = (const float*)d_in[11];
    const float* Wv  = (const float*)d_in[12];  const float* bv  = (const float*)d_in[13];
    const float* Wqg = (const float*)d_in[14];  const float* bqg = (const float*)d_in[15];
    const float* Wkg = (const float*)d_in[16];  const float* bkg = (const float*)d_in[17];
    const float* Wvg = (const float*)d_in[18];  const float* bvg = (const float*)d_in[19];
    const float* Wo  = (const float*)d_in[20];  const float* bo  = (const float*)d_in[21];
    const float* ln1g = (const float*)d_in[22]; const float* ln1b = (const float*)d_in[23];
    const float* Wf1 = (const float*)d_in[24];  const float* bf1 = (const float*)d_in[25];
    const float* Wf2 = (const float*)d_in[26];  const float* bf2 = (const float*)d_in[27];
    const float* ln2g = (const float*)d_in[28]; const float* ln2b = (const float*)d_in[29];
    const float* Wc  = (const float*)d_in[30];  const float* bc  = (const float*)d_in[31];
    const float* Wp  = (const float*)d_in[32];  const float* bp  = (const float*)d_in[33];

    const long M = (long)BB * SS;
    const float sc = 1.0f / sqrtf((float)DH);

    float* ws = (float*)d_ws;
    float* h     = ws;  ws += M * DD;
    float* qb    = ws;  ws += M * DD;
    float* kb    = ws;  ws += M * DD;
    float* vb    = ws;  ws += M * DD;
    float* kgb   = ws;  ws += M * DD;
    float* vgb   = ws;  ws += M * DD;
    float* attn  = ws;  ws += M * DD;
    float* tmpF  = ws;  ws += M * FF;
    float* tmp2  = ws;  ws += M * DD;
    float* qgbuf = ws;  ws += (long)BB * DD;
    float* tbuf  = ws;  ws += (long)BB * DD;

    embed_ln_kernel<<<dim3((unsigned)M), 256, 0, stream>>>(
        ids, tti, wordE, posE, typeE, ln_e_g, ln_e_b, h);

    for (int l = 0; l < LL; ++l) {
        const long wDD = (long)l * DD * DD;
        const long wDF = (long)l * DD * FF;
        const long vD  = (long)l * DD;
        const long vF  = (long)l * FF;

        launch_gemm(h, DD, Wq  + wDD, DD, bq  + vD, qb,  DD, (int)M, DD, DD, sc,  ACT_NONE, stream);
        launch_gemm(h, DD, Wk  + wDD, DD, bk  + vD, kb,  DD, (int)M, DD, DD, 1.f, ACT_NONE, stream);
        launch_gemm(h, DD, Wv  + wDD, DD, bv  + vD, vb,  DD, (int)M, DD, DD, 1.f, ACT_NONE, stream);
        launch_gemm(h, DD, Wkg + wDD, DD, bkg + vD, kgb, DD, (int)M, DD, DD, 1.f, ACT_NONE, stream);
        launch_gemm(h, DD, Wvg + wDD, DD, bvg + vD, vgb, DD, (int)M, DD, DD, 1.f, ACT_NONE, stream);
        launch_gemm(h, (long)SS * DD, Wqg + wDD, DD, bqg + vD, qgbuf, DD, BB, DD, DD, sc, ACT_NONE, stream);

        sliding_attn_kernel<<<dim3(SS / 16, HH, BB), 32, 0, stream>>>(qb, kb, vb, mask, attn);
        global_attn_kernel<<<dim3(HH, BB), 256, 0, stream>>>(qgbuf, kgb, vgb, mask, attn);

        launch_gemm(attn, DD, Wo + wDD, DD, bo + vD, tmp2, DD, (int)M, DD, DD, 1.f, ACT_NONE, stream);
        add_ln_kernel<<<dim3((unsigned)M), 256, 0, stream>>>(h, tmp2, ln1g + vD, ln1b + vD);

        launch_gemm(h,    DD, Wf1 + wDF, FF, bf1 + vF, tmpF, FF, (int)M, FF, DD, 1.f, ACT_GELU, stream);
        launch_gemm(tmpF, FF, Wf2 + wDF, DD, bf2 + vD, tmp2, DD, (int)M, DD, FF, 1.f, ACT_NONE, stream);
        add_ln_kernel<<<dim3((unsigned)M), 256, 0, stream>>>(h, tmp2, ln2g + vD, ln2b + vD);
    }

    launch_gemm(h, (long)SS * DD, Wc, DD, bc, tbuf, DD, BB, DD, DD, 1.f, ACT_TANH, stream);
    launch_gemm(tbuf, DD, Wp, NL, bp, (float*)d_out, NL, BB, NL, DD, 1.f, ACT_NONE, stream);
}
